// CausalSelfAttention_16630113370561
// MI455X (gfx1250) — compile-verified, matrix operand order checked by reading, not run on hardware
//
#include <hip/hip_runtime.h>
#include <hip/hip_bf16.h>

typedef __attribute__((ext_vector_type(16))) __bf16        v16bf;
typedef __attribute__((ext_vector_type(8)))  float         v8f;
typedef __attribute__((ext_vector_type(4)))  unsigned int  u32x4;

union Frag {
    v16bf v;
    u32x4 u[2];
};

// ---------------------------------------------------------------------------
// CDNA5 async global -> LDS copy (16B per lane), tracked by ASYNCcnt.
// ---------------------------------------------------------------------------
__device__ __forceinline__ void async_load_b128(unsigned lds_byte_off,
                                                const void* gptr) {
    asm volatile("global_load_async_to_lds_b128 %0, %1, off"
                 :: "v"(lds_byte_off), "v"(gptr) : "memory");
}
__device__ __forceinline__ void wait_asynccnt0() {
    asm volatile("s_wait_asynccnt 0x0" ::: "memory");
}

// ---------------------------------------------------------------------------
// fp32 -> bf16 elementwise conversion
// ---------------------------------------------------------------------------
__global__ void convert_f32_bf16(const float* __restrict__ in,
                                 __bf16* __restrict__ out, int n) {
    int i = blockIdx.x * blockDim.x + threadIdx.x;
    if (i < n) out[i] = (__bf16)in[i];
}

// ---------------------------------------------------------------------------
// C[m,n] = sum_k A[m,k] * W[n,k]   (A: MxK bf16, W: NxK bf16, C: MxN f32)
// Block: 256 threads (8 waves) -> 64x128 output tile.
//   Wave w: (wm = w&1, wn = w>>1) -> 32x32 quad of 16x16 WMMA tiles.
// Per k-step of 32: stage A(64x32) + W(128x32) to LDS with
// global_load_async_to_lds_b128, wait ASYNCcnt, barrier, then 8x ds_load_b128
// fragments feed 4 v_wmma_f32_16x16x32_bf16 per wave (2x operand reuse).
// Fragment maps (wave32): A lane m=L&15, half=L>>4,
//   elem i -> k = (i<8) ? half*8+i : 16+half*8+(i-8); B mirrors with n=L&15.
// C/D: row = r + 8*half, col = L&15.
// ---------------------------------------------------------------------------
__global__ void gemm_bf16_wmma(const __bf16* __restrict__ A,
                               const __bf16* __restrict__ W,
                               float* __restrict__ C,
                               int M, int N, int K) {
    __shared__ __align__(16) __bf16 Atile[64 * 32];    // 4 KB
    __shared__ __align__(16) __bf16 Btile[128 * 32];   // 8 KB

    const int tid  = threadIdx.x;
    const int lane = tid & 31;
    const int wave = tid >> 5;
    const int wm   = wave & 1;        // 0..1 -> M offset 0/32
    const int wn   = wave >> 1;       // 0..3 -> N offset 0/32/64/96
    const int half = lane >> 4;
    const int mn   = lane & 15;

    const int m0 = blockIdx.x * 64;
    const int n0 = blockIdx.y * 128;

    const unsigned aBase = (unsigned)(uintptr_t)Atile;
    const unsigned bBase = (unsigned)(uintptr_t)Btile;

    // Staging assignment: 16B segments. A: 256 segs (1/thread), B: 512 (2/thread).
    const int arow = tid >> 2, aseg = tid & 3;
    const unsigned aLds  = aBase + (unsigned)(arow * 32 + aseg * 8) * 2;
    const __bf16*  aG    = A + (size_t)(m0 + arow) * K + aseg * 8;
    const int brow = tid >> 2, bseg = tid & 3;
    const unsigned bLds0 = bBase + (unsigned)(brow * 32 + bseg * 8) * 2;
    const unsigned bLds1 = bBase + (unsigned)((brow + 64) * 32 + bseg * 8) * 2;
    const __bf16*  bG0   = W + (size_t)(n0 + brow) * K + bseg * 8;
    const __bf16*  bG1   = W + (size_t)(n0 + brow + 64) * K + bseg * 8;

    v8f acc[2][2] = {{v8f{}, v8f{}}, {v8f{}, v8f{}}};

    for (int k0 = 0; k0 < K; k0 += 32) {
        // ---- async stage this k-slice into LDS -----------------------------
        async_load_b128(aLds,  aG  + k0);
        async_load_b128(bLds0, bG0 + k0);
        async_load_b128(bLds1, bG1 + k0);
        wait_asynccnt0();
        __syncthreads();

        // ---- fragments from LDS -------------------------------------------
        Frag a[2], b[2];
#pragma unroll
        for (int i = 0; i < 2; ++i) {
            const __bf16* ar = &Atile[(wm * 32 + i * 16 + mn) * 32];
            a[i].u[0] = *(const u32x4*)(ar + half * 8);
            a[i].u[1] = *(const u32x4*)(ar + 16 + half * 8);
            const __bf16* br = &Btile[(wn * 32 + i * 16 + mn) * 32];
            b[i].u[0] = *(const u32x4*)(br + half * 8);
            b[i].u[1] = *(const u32x4*)(br + 16 + half * 8);
        }
#pragma unroll
        for (int i = 0; i < 2; ++i)
#pragma unroll
            for (int j = 0; j < 2; ++j)
                acc[i][j] = __builtin_amdgcn_wmma_f32_16x16x32_bf16(
                    false, a[i].v, false, b[j].v, (short)0, acc[i][j],
                    false, false);
        __syncthreads();   // protect LDS before next k-slice overwrites it
    }

    // ---- store 2x2 quad of 16x16 tiles ------------------------------------
#pragma unroll
    for (int i = 0; i < 2; ++i)
#pragma unroll
        for (int j = 0; j < 2; ++j) {
            float* out = C + (size_t)(m0 + wm * 32 + i * 16) * N +
                         (size_t)(n0 + wn * 32 + j * 16) + mn;
#pragma unroll
            for (int r = 0; r < 8; ++r)
                out[(size_t)(r + 8 * half) * N] = acc[i][j][r];
        }
}

// ---------------------------------------------------------------------------
// RoPE on q,k + scatter to attention layouts (bf16).
//   qkv  : [B*T, 3*C] f32   (C = 1024, H = 16, D = 64)
//   Qb,Kb: [B,H,T,D] bf16 (RoPE applied)
//   Vt   : [B,H,D,T] bf16 (transposed for contiguous-K PV fragments)
// ---------------------------------------------------------------------------
__global__ void rope_scatter(const float* __restrict__ qkv,
                             const float* __restrict__ cosf,
                             const float* __restrict__ sinf,
                             __bf16* __restrict__ Qb,
                             __bf16* __restrict__ Kb,
                             __bf16* __restrict__ Vt) {
    const int T = 2048, H = 16, D = 64;
    int idx = blockIdx.x * blockDim.x + threadIdx.x;   // 2^21 threads
    int j = idx & 31;
    int h = (idx >> 5) & 15;
    int t = (idx >> 9) & 2047;
    int b = idx >> 20;

    const float* row = qkv + (size_t)(b * T + t) * 3072;
    float c = cosf[t * 32 + j];
    float s = sinf[t * 32 + j];
    int d0 = 2 * j;

    float q0 = row[h * D + d0],        q1 = row[h * D + d0 + 1];
    float k0 = row[1024 + h * D + d0], k1 = row[1024 + h * D + d0 + 1];
    float v0 = row[2048 + h * D + d0], v1 = row[2048 + h * D + d0 + 1];

    size_t base = (((size_t)b * H + h) * T + t) * D + d0;
    Qb[base]     = (__bf16)(q0 * c - q1 * s);
    Qb[base + 1] = (__bf16)(q0 * s + q1 * c);
    Kb[base]     = (__bf16)(k0 * c - k1 * s);
    Kb[base + 1] = (__bf16)(k0 * s + k1 * c);

    size_t vbase = (((size_t)b * H + h) * D + d0) * T + t;
    Vt[vbase]     = (__bf16)v0;
    Vt[vbase + T] = (__bf16)v1;
}

// ---------------------------------------------------------------------------
// Causal flash attention, one wave per (b,h, 16-query tile).
//   S  = Q K^T * 1/8  (two chained wmma per 16-key tile, Dk=64)
//   online softmax in C/D layout (row r+8*half; shfl_xor over 16-lane half)
//   P  -> LDS -> A-fragment, O += P V (V from transposed Vt)
// Output: attn [B*T, H*D] bf16 (ready as GEMM A for projection).
// ---------------------------------------------------------------------------
__global__ void flash_attn(const __bf16* __restrict__ Qb,
                           const __bf16* __restrict__ Kb,
                           const __bf16* __restrict__ Vt,
                           __bf16* __restrict__ attn) {
    const int T = 2048, H = 16, D = 64;
    __shared__ __align__(32) __bf16 Plds[16 * 32];

    const int qt   = blockIdx.x;          // query tile (0..127)
    const int bh   = blockIdx.y;          // b*H + h    (0..31)
    const int lane = threadIdx.x;
    const int half = lane >> 4;
    const int mn   = lane & 15;
    const int q0   = qt * 16;
    const float scale = 0.125f;           // 1/sqrt(64)

    const __bf16* Qbase = Qb + (size_t)bh * T * D;
    const __bf16* Kbase = Kb + (size_t)bh * T * D;
    const __bf16* Vbase = Vt + (size_t)bh * D * T;

    // Q fragments (rows q0+mn), two 32-wide K chunks of Dk=64
    Frag qa0, qa1;
    {
        const __bf16* qrow = Qbase + (size_t)(q0 + mn) * D;
        qa0.u[0] = *(const u32x4*)(qrow + half * 8);
        qa0.u[1] = *(const u32x4*)(qrow + 16 + half * 8);
        qa1.u[0] = *(const u32x4*)(qrow + 32 + half * 8);
        qa1.u[1] = *(const u32x4*)(qrow + 48 + half * 8);
    }

    v8f o[4] = {v8f{}, v8f{}, v8f{}, v8f{}};
    float mrow[8], lrow[8];
#pragma unroll
    for (int r = 0; r < 8; ++r) { mrow[r] = -3.0e38f; lrow[r] = 0.0f; }

    const int nblocks = (q0 + 16 + 31) / 32;   // causal: key blocks of 32
    for (int jb = 0; jb < nblocks; ++jb) {
        const int kbase = jb * 32;

        // ---- S = Q K^T for two 16-key tiles --------------------------------
        v8f s[2];
#pragma unroll
        for (int tile = 0; tile < 2; ++tile) {
            const __bf16* krow = Kbase + (size_t)(kbase + tile * 16 + mn) * D;
            Frag kb0, kb1;
            kb0.u[0] = *(const u32x4*)(krow + half * 8);
            kb0.u[1] = *(const u32x4*)(krow + 16 + half * 8);
            kb1.u[0] = *(const u32x4*)(krow + 32 + half * 8);
            kb1.u[1] = *(const u32x4*)(krow + 48 + half * 8);
            v8f st = {};
            st = __builtin_amdgcn_wmma_f32_16x16x32_bf16(
                false, qa0.v, false, kb0.v, (short)0, st, false, false);
            st = __builtin_amdgcn_wmma_f32_16x16x32_bf16(
                false, qa1.v, false, kb1.v, (short)0, st, false, false);
            s[tile] = st;
        }

        // ---- scale + causal mask + online softmax --------------------------
        float alpha[8];
#pragma unroll
        for (int r = 0; r < 8; ++r) {
            const int qi = q0 + r + 8 * half;
            float v0 = (kbase + mn      <= qi) ? s[0][r] * scale : -3.0e38f;
            float v1 = (kbase + 16 + mn <= qi) ? s[1][r] * scale : -3.0e38f;
            float mx = fmaxf(v0, v1);
#pragma unroll
            for (int off = 1; off < 16; off <<= 1)
                mx = fmaxf(mx, __shfl_xor(mx, off, 32));
            const float newm = fmaxf(mrow[r], mx);
            alpha[r] = __expf(mrow[r] - newm);
            mrow[r] = newm;

            float p0 = __expf(v0 - newm);
            float p1 = __expf(v1 - newm);
            float ps = p0 + p1;
#pragma unroll
            for (int off = 1; off < 16; off <<= 1)
                ps += __shfl_xor(ps, off, 32);
            lrow[r] = lrow[r] * alpha[r] + ps;
            s[0][r] = p0;
            s[1][r] = p1;
        }

        // ---- P: C/D layout -> LDS -> A-fragment layout ---------------------
        __syncthreads();
#pragma unroll
        for (int tile = 0; tile < 2; ++tile)
#pragma unroll
            for (int r = 0; r < 8; ++r)
                Plds[(r + 8 * half) * 32 + tile * 16 + mn] = (__bf16)s[tile][r];
        __syncthreads();

        Frag pa;
        pa.u[0] = *(const u32x4*)(&Plds[mn * 32 + half * 8]);
        pa.u[1] = *(const u32x4*)(&Plds[mn * 32 + 16 + half * 8]);

        // ---- O = O*alpha + P V ---------------------------------------------
#pragma unroll
        for (int c = 0; c < 4; ++c) {
#pragma unroll
            for (int r = 0; r < 8; ++r) o[c][r] *= alpha[r];
            const __bf16* vcol = Vbase + (size_t)(c * 16 + mn) * T + kbase;
            Frag vb;
            vb.u[0] = *(const u32x4*)(vcol + half * 8);
            vb.u[1] = *(const u32x4*)(vcol + 16 + half * 8);
            o[c] = __builtin_amdgcn_wmma_f32_16x16x32_bf16(
                false, pa.v, false, vb.v, (short)0, o[c], false, false);
        }
    }

    // ---- epilogue: divide by row sum, store bf16 [B*T, H*D] ----------------
    const int b = bh >> 4, h = bh & 15;
#pragma unroll
    for (int c = 0; c < 4; ++c) {
#pragma unroll
        for (int r = 0; r < 8; ++r) {
            const int rowm = r + 8 * half;
            const float inv = 1.0f / lrow[r];
            size_t off = (size_t)(b * T + q0 + rowm) * 1024 +
                         (size_t)h * D + c * 16 + mn;
            attn[off] = (__bf16)(o[c][r] * inv);
        }
    }
}

// ---------------------------------------------------------------------------
// Host-side launcher
// ---------------------------------------------------------------------------
extern "C" void kernel_launch(void* const* d_in, const int* in_sizes, int n_in,
                              void* d_out, int out_size, void* d_ws, size_t ws_size,
                              hipStream_t stream) {
    (void)in_sizes; (void)n_in; (void)out_size; (void)ws_size;
    const int B = 2, T = 2048, C = 1024, H = 16, D = 64;
    const int M = B * T;           // 4096

    const float* x      = (const float*)d_in[0];
    const float* fcos   = (const float*)d_in[1];
    const float* fsin   = (const float*)d_in[2];
    const float* qkv_w  = (const float*)d_in[3];
    const float* proj_w = (const float*)d_in[4];
    float* out = (float*)d_out;

    char* ws = (char*)d_ws;
    size_t off = 0;
    auto alloc = [&](size_t bytes) {
        void* p = ws + off;
        off += (bytes + 255) & ~(size_t)255;
        return p;
    };
    __bf16* xb    = (__bf16*)alloc((size_t)M * C * 2);          //  8 MiB
    __bf16* qwb   = (__bf16*)alloc((size_t)3 * C * C * 2);      //  6 MiB
    __bf16* pwb   = (__bf16*)alloc((size_t)C * C * 2);          //  2 MiB
    float*  qkvf  = (float*)alloc((size_t)M * 3 * C * 4);       // 48 MiB
    __bf16* Qb    = (__bf16*)alloc((size_t)M * C * 2);          //  8 MiB
    __bf16* Kb    = (__bf16*)alloc((size_t)M * C * 2);          //  8 MiB
    __bf16* Vt    = (__bf16*)alloc((size_t)M * C * 2);          //  8 MiB
    __bf16* attnb = (__bf16*)alloc((size_t)M * C * 2);          //  8 MiB

    // 1) fp32 -> bf16
    {
        int n1 = M * C, n2 = 3 * C * C, n3 = C * C;
        convert_f32_bf16<<<(n1 + 255) / 256, 256, 0, stream>>>(x, xb, n1);
        convert_f32_bf16<<<(n2 + 255) / 256, 256, 0, stream>>>(qkv_w, qwb, n2);
        convert_f32_bf16<<<(n3 + 255) / 256, 256, 0, stream>>>(proj_w, pwb, n3);
    }

    // 2) qkv = x @ qkv_w^T   (M=4096, N=3072, K=1024), 64x128 block tiles
    gemm_bf16_wmma<<<dim3(M / 64, (3 * C) / 128), 256, 0, stream>>>(
        xb, qwb, qkvf, M, 3 * C, C);

    // 3) RoPE + scatter to [B,H,T,D] (q,k) and [B,H,D,T] (v)
    rope_scatter<<<(B * T * H * (D / 2)) / 256, 256, 0, stream>>>(
        qkvf, fcos, fsin, Qb, Kb, Vt);

    // 4) causal flash attention -> attn [B*T, H*D] bf16
    flash_attn<<<dim3(T / 16, B * H), 32, 0, stream>>>(Qb, Kb, Vt, attnb);

    // 5) out = attn @ proj_w^T   (M=4096, N=1024, K=1024), fp32 output
    gemm_bf16_wmma<<<dim3(M / 64, C / 128), 256, 0, stream>>>(
        attnb, pwb, out, M, C, C);
}
